// SelfAttention_24799141167268
// MI455X (gfx1250) — matrix operand order repaired and checked by reading, not compiled, not run on hardware
//
#include <hip/hip_runtime.h>
#include <hip/hip_bf16.h>
#include <math.h>

// ---------------- problem constants ----------------
#define BATCH 2
#define SEQ   2048
#define DIM   1024
#define HEADS 16
#define HD    64
#define EPS   1.1920928955078125e-07f

typedef __bf16 bf16_t;
typedef __attribute__((ext_vector_type(16))) __bf16 v16bf;
typedef __attribute__((ext_vector_type(8)))  __bf16 v8bf;
typedef __attribute__((ext_vector_type(8)))  float  v8f;
typedef __attribute__((ext_vector_type(4)))  float  v4f;

// ---------------- WMMA fragment loaders ----------------
// A operand (16x32 bf16, MxK): lane l holds row (l&15);
// lanes<16: K = 0..15 contiguous; lanes>=16: K = 16..31 contiguous
__device__ __forceinline__ v16bf load_a_frag(const bf16_t* __restrict__ p, int lane) {
  const int koff = (lane & 16) ? 16 : 0;
  v8bf lo = *(const v8bf*)(p + koff);
  v8bf hi = *(const v8bf*)(p + koff + 8);
  v16bf r;
#pragma unroll
  for (int e = 0; e < 8; ++e) { r[e] = lo[e]; r[e + 8] = hi[e]; }
  return r;
}

// B operand (32x16 bf16, KxN): lane n holds column (n&15);
// lanes<16: K = 0..15 contiguous; lanes>=16: K = 16..31 contiguous
__device__ __forceinline__ v16bf load_b_frag(const bf16_t* __restrict__ p, int lane) {
  const int koff = (lane & 16) ? 16 : 0;
  return *(const v16bf*)(p + koff);
}

__device__ __forceinline__ v8f wmma_bf16(v16bf a, v16bf b, v8f c) {
  return __builtin_amdgcn_wmma_f32_16x16x32_bf16(false, a, false, b, (short)0, c,
                                                 false, false);
}

// ---------------- CDNA5 async global->LDS copy (ASYNCcnt path) ----------------
// 16-byte DMA from global memory directly into LDS, no VGPR round trip.
// LDS address = low 32 bits of the flat shared pointer (ISA 10.2: LDS aperture
// keeps the LDS byte offset in addr[31:0]).
__device__ __forceinline__ void async_copy_b128(const void* gptr, void* lptr) {
  const unsigned lds_addr = (unsigned)(unsigned long long)(uintptr_t)lptr;
  const unsigned long long gaddr = (unsigned long long)(uintptr_t)gptr;
  asm volatile("global_load_async_to_lds_b128 %0, %1, off"
               :
               : "v"(lds_addr), "v"(gaddr)
               : "memory");
}

__device__ __forceinline__ void async_wait_all() {
  asm volatile("s_wait_asynccnt 0x0" ::: "memory");
}

// ---------------- fp32 -> bf16 convert ----------------
__global__ __launch_bounds__(256) void cvt_f32_bf16(const float* __restrict__ in,
                                                    bf16_t* __restrict__ out, int n4) {
  int i = blockIdx.x * blockDim.x + threadIdx.x;
  if (i >= n4) return;
  v4f v = ((const v4f*)in)[i];
#pragma unroll
  for (int j = 0; j < 4; ++j) out[4 * i + j] = (bf16_t)v[j];
}

// ---------------- WMMA GEMM: out[M,N] = A[M,K] @ W[N,K]^T + bias ----------------
// block = 256 threads (8 waves), block tile 128x128, wave tile 32x64
__global__ __launch_bounds__(256) void gemm_bf16_wmma(const bf16_t* __restrict__ A,
                                                      const bf16_t* __restrict__ W,
                                                      const float* __restrict__ bias,
                                                      float* __restrict__ out,
                                                      int M, int N, int K) {
  const int lane = threadIdx.x & 31;
  const int wave = threadIdx.x >> 5;
  const int wm = wave & 3;   // 4 waves along M (32 rows each)
  const int wn = wave >> 2;  // 2 waves along N (64 cols each)
  const int m0 = blockIdx.y * 128 + wm * 32;
  const int n0 = blockIdx.x * 128 + wn * 64;

  v8f acc[2][4];
  const v8f zero = {0.f, 0.f, 0.f, 0.f, 0.f, 0.f, 0.f, 0.f};
#pragma unroll
  for (int mi = 0; mi < 2; ++mi)
#pragma unroll
    for (int ni = 0; ni < 4; ++ni) acc[mi][ni] = zero;

  const bf16_t* arow[2];
  const bf16_t* wrow[4];
#pragma unroll
  for (int mi = 0; mi < 2; ++mi)
    arow[mi] = A + (size_t)(m0 + mi * 16 + (lane & 15)) * K;
#pragma unroll
  for (int ni = 0; ni < 4; ++ni)
    wrow[ni] = W + (size_t)(n0 + ni * 16 + (lane & 15)) * K;

  for (int k = 0; k < K; k += 32) {
    v16bf af[2], bf[4];
#pragma unroll
    for (int mi = 0; mi < 2; ++mi) af[mi] = load_a_frag(arow[mi] + k, lane);
#pragma unroll
    for (int ni = 0; ni < 4; ++ni) bf[ni] = load_b_frag(wrow[ni] + k, lane);
#pragma unroll
    for (int mi = 0; mi < 2; ++mi)
#pragma unroll
      for (int ni = 0; ni < 4; ++ni) acc[mi][ni] = wmma_bf16(af[mi], bf[ni], acc[mi][ni]);
  }

  const int rbase = (lane & 16) ? 8 : 0;
#pragma unroll
  for (int ni = 0; ni < 4; ++ni) {
    const int col = n0 + ni * 16 + (lane & 15);
    const float bv = bias[col];
#pragma unroll
    for (int mi = 0; mi < 2; ++mi) {
#pragma unroll
      for (int r = 0; r < 8; ++r) {
        out[(size_t)(m0 + mi * 16 + rbase + r) * N + col] = acc[mi][ni][r] + bv;
      }
    }
  }
}

// ---------------- RMSNorm + RoPE + pack Q/K/Vt as bf16 ----------------
// one thread per (b,h,l) row of 64 elements
__global__ __launch_bounds__(256) void prep_qkv(const float* __restrict__ qkv,
                                                const float* __restrict__ pe,
                                                const float* __restrict__ qw,
                                                const float* __restrict__ kw,
                                                bf16_t* __restrict__ Qb,
                                                bf16_t* __restrict__ Kb,
                                                bf16_t* __restrict__ Vt) {
  const int tid = blockIdx.x * blockDim.x + threadIdx.x;  // 0 .. B*H*L-1
  const int l  = tid & (SEQ - 1);
  const int bh = tid >> 11;
  const int h  = bh & (HEADS - 1);

  const float* base = qkv + (size_t)((bh >> 4) * SEQ + l) * (3 * DIM);
  const float* perow = pe + (size_t)l * (HD / 2) * 4;

#pragma unroll
  for (int s = 0; s < 2; ++s) {
    const float* src = base + s * DIM + h * HD;
    const float* w = s ? kw : qw;
    float t[HD];
    float ss = 0.f;
#pragma unroll
    for (int d4 = 0; d4 < HD / 4; ++d4) {
      v4f v = ((const v4f*)src)[d4];
#pragma unroll
      for (int j = 0; j < 4; ++j) {
        t[4 * d4 + j] = v[j];
        ss += v[j] * v[j];
      }
    }
    const float rinv = rsqrtf(ss * (1.0f / HD) + EPS);
    const float sc = s ? 1.0f : 0.125f;  // fold 1/sqrt(HD) into q
    bf16_t* dst = (s ? Kb : Qb) + ((size_t)bh * SEQ + l) * HD;
#pragma unroll
    for (int dh = 0; dh < HD / 2; ++dh) {
      const float a = t[2 * dh] * rinv * w[2 * dh];
      const float b = t[2 * dh + 1] * rinv * w[2 * dh + 1];
      const float* pp = perow + dh * 4;
      dst[2 * dh]     = (bf16_t)((pp[0] * a + pp[1] * b) * sc);
      dst[2 * dh + 1] = (bf16_t)((pp[2] * a + pp[3] * b) * sc);
    }
  }

  // V: transpose to [64][SEQ] per (b,h)
  const float* vsrc = base + 2 * DIM + h * HD;
  bf16_t* vd = Vt + (size_t)bh * HD * SEQ + l;
#pragma unroll
  for (int d = 0; d < HD; ++d) vd[(size_t)d * SEQ] = (bf16_t)vsrc[d];
}

// ---------------- flash attention (async-LDS staged K/V) ----------------
// grid (SEQ/128, B*H); 8 waves; wave = 16 query rows; KV blocks of 32.
// K/V blocks are async-DMA'd to double-buffered LDS once per workgroup
// (instead of 8x redundantly per wave), then consumed as WMMA fragments
// via ds loads. Computes S^T = K @ Q^T so P repacks into the A-operand
// layout entirely in-lane.
__global__ __launch_bounds__(256) void flash_attn(const bf16_t* __restrict__ Qb,
                                                  const bf16_t* __restrict__ Kb,
                                                  const bf16_t* __restrict__ Vt,
                                                  bf16_t* __restrict__ Ob) {
  __shared__ __align__(64) bf16_t Ks[2][32 * HD];  // [kv 32][d 64], 4KB x2
  __shared__ __align__(64) bf16_t Vs[2][HD * 32];  // [d 64][kv 32], 4KB x2

  const int lane = threadIdx.x & 31;
  const int wave = threadIdx.x >> 5;
  const int tid = threadIdx.x;
  const int bh = blockIdx.y;
  const int b = bh >> 4, h = bh & (HEADS - 1);
  const int q0 = blockIdx.x * 128 + wave * 16;

  const bf16_t* Qp = Qb + (size_t)bh * SEQ * HD;
  const bf16_t* Kp = Kb + (size_t)bh * SEQ * HD;
  const bf16_t* Vp = Vt + (size_t)bh * HD * SEQ;

  // Q as B operand: lane n holds query row q0+(n&15); 2 k-steps over HD=64
  v16bf qf[2];
  {
    const bf16_t* qrow = Qp + (size_t)(q0 + (lane & 15)) * HD;
    const int koff = (lane & 16) ? 16 : 0;
    qf[0] = *(const v16bf*)(qrow + koff);
    qf[1] = *(const v16bf*)(qrow + 32 + koff);
  }

  const v8f zero = {0.f, 0.f, 0.f, 0.f, 0.f, 0.f, 0.f, 0.f};
  v8f o[4];
#pragma unroll
  for (int ni = 0; ni < 4; ++ni) o[ni] = zero;
  float m = -__builtin_inff();
  float lsum = 0.f;
  const int rb = (lane & 16) ? 8 : 0;

  // stage KV block j0 into LDS buffer `buf`: one b128 per thread per tile
  auto stage = [&](int buf, int j0) {
    // K tile: row = tid>>3 (32 rows), chunk = tid&7; LDS row-major [32][64]
    async_copy_b128(Kp + (size_t)(j0 + (tid >> 3)) * HD + (tid & 7) * 8,
                    &Ks[buf][tid * 8]);
    // V tile: row d = tid>>2 (64 rows), chunk = tid&3; LDS [64][32]
    async_copy_b128(Vp + (size_t)(tid >> 2) * SEQ + j0 + (tid & 3) * 8,
                    &Vs[buf][tid * 8]);
  };

  stage(0, 0);
  int ib = 0;
  for (int j0 = 0; j0 < SEQ; j0 += 32) {
    async_wait_all();   // my async writes to buffer ib have landed in LDS
    __syncthreads();    // everyone's have landed
    if (j0 + 32 < SEQ) stage(ib ^ 1, j0 + 32);  // overlap DMA with compute

    // S^T tiles: sub 0 -> kv rows j0..j0+15, sub 1 -> j0+16..j0+31
    v8f st[2];
#pragma unroll
    for (int sub = 0; sub < 2; ++sub) {
      const bf16_t* krow = &Ks[ib][(sub * 16 + (lane & 15)) * HD];
      v16bf ka0 = load_a_frag(krow, lane);
      v16bf ka1 = load_a_frag(krow + 32, lane);
      v8f c = zero;
      c = wmma_bf16(ka0, qf[0], c);
      c = wmma_bf16(ka1, qf[1], c);
      st[sub] = c;
    }

    // online softmax over this lane's 16 scores of query row (lane&15)
    float mloc = st[0][0];
#pragma unroll
    for (int r = 0; r < 8; ++r) {
      mloc = fmaxf(mloc, st[0][r]);
      mloc = fmaxf(mloc, st[1][r]);
    }
    mloc = fmaxf(mloc, __shfl_xor(mloc, 16, 32));
    const float mnew = fmaxf(m, mloc);
    const float alpha = __expf(m - mnew);

    float p0[8], p1[8], psum = 0.f;
#pragma unroll
    for (int r = 0; r < 8; ++r) {
      p0[r] = __expf(st[0][r] - mnew);
      p1[r] = __expf(st[1][r] - mnew);
      psum += p0[r] + p1[r];
    }
    psum += __shfl_xor(psum, 16, 32);
    lsum = lsum * alpha + psum;
    m = mnew;

    // P -> A-operand fragment: pure in-lane repack (layout match by design)
    v16bf pf;
#pragma unroll
    for (int e = 0; e < 8; ++e) {
      pf[e] = (bf16_t)p0[e];
      pf[e + 8] = (bf16_t)p1[e];
    }

    // rescale O accumulators by per-row alpha (broadcast rows rb..rb+7)
    float asc[8];
#pragma unroll
    for (int r = 0; r < 8; ++r) asc[r] = __shfl(alpha, rb + r, 32);
#pragma unroll
    for (int ni = 0; ni < 4; ++ni)
#pragma unroll
      for (int r = 0; r < 8; ++r) o[ni][r] *= asc[r];

    // O += P @ V  (V^T rows contiguous along kv in LDS)
#pragma unroll
    for (int ni = 0; ni < 4; ++ni) {
      const bf16_t* vrow =
          &Vs[ib][(ni * 16 + (lane & 15)) * 32 + ((lane & 16) ? 8 : 0)];
      v8bf vlo = *(const v8bf*)vrow;
      v8bf vhi = *(const v8bf*)(vrow + 16);
      v16bf vf;
#pragma unroll
      for (int e = 0; e < 8; ++e) { vf[e] = vlo[e]; vf[e + 8] = vhi[e]; }
      o[ni] = wmma_bf16(pf, vf, o[ni]);
    }
    ib ^= 1;
  }

  // normalize and store to Ob[B,L,DIM] (token-major for the proj GEMM)
  float inv[8];
#pragma unroll
  for (int r = 0; r < 8; ++r) inv[r] = 1.0f / __shfl(lsum, rb + r, 32);
#pragma unroll
  for (int ni = 0; ni < 4; ++ni) {
    const int d = h * HD + ni * 16 + (lane & 15);
#pragma unroll
    for (int r = 0; r < 8; ++r) {
      const int lq = q0 + rb + r;
      Ob[(size_t)(b * SEQ + lq) * DIM + d] = (bf16_t)(o[ni][r] * inv[r]);
    }
  }
}

// ---------------- host launcher ----------------
extern "C" void kernel_launch(void* const* d_in, const int* in_sizes, int n_in,
                              void* d_out, int out_size, void* d_ws, size_t ws_size,
                              hipStream_t stream) {
  const float* x       = (const float*)d_in[0];
  const float* pe      = (const float*)d_in[1];
  const float* qkv_w   = (const float*)d_in[2];
  const float* qkv_b   = (const float*)d_in[3];
  const float* qnw     = (const float*)d_in[4];
  const float* knw     = (const float*)d_in[5];
  const float* proj_w  = (const float*)d_in[6];
  const float* proj_b  = (const float*)d_in[7];
  float* out = (float*)d_out;
  char* ws = (char*)d_ws;

  const int M = BATCH * SEQ;  // 4096

  // workspace layout (bytes)
  bf16_t* xb    = (bf16_t*)(ws + 0);                    //  8 MB  x as bf16
  bf16_t* wqkv  = (bf16_t*)(ws + 8388608);              //  6 MB  qkv_w bf16
  bf16_t* wproj = (bf16_t*)(ws + 14680064);             //  2 MB  proj_w bf16
  float*  qkv   = (float*)(ws + 16777216);              // 48 MB  qkv fp32
  bf16_t* Qb    = (bf16_t*)(ws + 67108864);             //  8 MB
  bf16_t* Kb    = (bf16_t*)(ws + 75497472);             //  8 MB
  bf16_t* Vt    = (bf16_t*)(ws + 83886080);             //  8 MB
  bf16_t* Ob    = (bf16_t*)(ws + 92274688);             //  8 MB

  // 1) fp32 -> bf16 converts
  {
    int n4 = M * DIM / 4;
    cvt_f32_bf16<<<(n4 + 255) / 256, 256, 0, stream>>>(x, xb, n4);
    n4 = 3 * DIM * DIM / 4;
    cvt_f32_bf16<<<(n4 + 255) / 256, 256, 0, stream>>>(qkv_w, wqkv, n4);
    n4 = DIM * DIM / 4;
    cvt_f32_bf16<<<(n4 + 255) / 256, 256, 0, stream>>>(proj_w, wproj, n4);
  }

  // 2) qkv = x @ qkv_w^T + qkv_b
  {
    dim3 grid(3 * DIM / 128, M / 128);
    gemm_bf16_wmma<<<grid, 256, 0, stream>>>(xb, wqkv, qkv_b, qkv, M, 3 * DIM, DIM);
  }

  // 3) RMSNorm + RoPE + pack
  {
    const int rows = BATCH * HEADS * SEQ;  // 65536
    prep_qkv<<<rows / 256, 256, 0, stream>>>(qkv, pe, qnw, knw, Qb, Kb, Vt);
  }

  // 4) flash attention
  {
    dim3 grid(SEQ / 128, BATCH * HEADS);
    flash_attn<<<grid, 256, 0, stream>>>(Qb, Kb, Vt, Ob);
  }

  // 5) out = Ob @ proj_w^T + proj_b
  {
    dim3 grid(DIM / 128, M / 128);
    gemm_bf16_wmma<<<grid, 256, 0, stream>>>(Ob, wproj, proj_b, out, M, DIM, DIM);
  }
}
